// CrossAttention_74586402062774
// MI455X (gfx1250) — compile-verified, matrix operand order checked by reading, not run on hardware
//
#include <hip/hip_runtime.h>
#include <hip/hip_bf16.h>

typedef __bf16 bf16;
typedef __attribute__((ext_vector_type(16))) __bf16 v16bf;
typedef __attribute__((ext_vector_type(8)))  float  v8f;
typedef __attribute__((ext_vector_type(4)))  unsigned int u32x4;
typedef __attribute__((ext_vector_type(8)))  int i32x8;
typedef __attribute__((ext_vector_type(4)))  int i32x4;

#ifndef __has_builtin
#define __has_builtin(x) 0
#endif
#if defined(__AMDGCN__) && __has_builtin(__builtin_amdgcn_tensor_load_to_lds)
#define ATHENA_TDM 1
#else
#define ATHENA_TDM 0
#endif

// ---------------------------------------------------------------------------
// Fragment helpers.
// 16-bit A-matrix 16x32 layout (CDNA5 ISA):
//   lane L: row M = L&15, half = L>>4
//   elements 0..7  hold K = 8*half + 0..7      (16 contiguous bytes)
//   elements 8..15 hold K = 16 + 8*half + 0..7 (16 contiguous bytes)
// B fragments mirror this with lane = output column, elements = K.
// ---------------------------------------------------------------------------
union FragBF { v16bf v; uint4 q[2]; };

__device__ __forceinline__ v16bf ld_frag(const bf16* row, int half) {
  FragBF f;
  f.q[0] = *reinterpret_cast<const uint4*>(row + 8 * half);
  f.q[1] = *reinterpret_cast<const uint4*>(row + 16 + 8 * half);
  return f.v;
}

__device__ __forceinline__ v8f wmma_bf16(v16bf a, v16bf b, v8f c) {
  return __builtin_amdgcn_wmma_f32_16x16x32_bf16(false, a, false, b, (short)0, c,
                                                 false, false);
}

__device__ __forceinline__ v8f v8f_zero() {
  v8f z;
#pragma unroll
  for (int i = 0; i < 8; ++i) z[i] = 0.f;
  return z;
}

// ---------------------------------------------------------------------------
// Cross-lane reductions on the VALU (v_permlane16_b32 / v_permlanex16_b32).
// ---------------------------------------------------------------------------
constexpr unsigned sel_lo(int m) {
  unsigned v = 0;
  for (int i = 0; i < 8; ++i) v |= (unsigned)((i ^ m) & 15) << (4 * i);
  return v;
}
constexpr unsigned sel_hi(int m) {
  unsigned v = 0;
  for (int i = 8; i < 16; ++i) v |= (unsigned)((i ^ m) & 15) << (4 * (i - 8));
  return v;
}

template <int M>
__device__ __forceinline__ float permxor16(float v) {
  constexpr unsigned lo = sel_lo(M), hi = sel_hi(M);
  int x = __builtin_bit_cast(int, v);
  x = __builtin_amdgcn_permlane16(x, x, (int)lo, (int)hi, true, false);
  return __builtin_bit_cast(float, x);
}

__device__ __forceinline__ float permswap_half(float v) {
  int x = __builtin_bit_cast(int, v);
  x = __builtin_amdgcn_permlanex16(x, x, (int)0x76543210u, (int)0xFEDCBA98u,
                                   true, false);
  return __builtin_bit_cast(float, x);
}

__device__ __forceinline__ float red_add16(float v) {
  v += permxor16<1>(v);
  v += permxor16<2>(v);
  v += permxor16<4>(v);
  v += permxor16<8>(v);
  return v;
}
__device__ __forceinline__ float red_max16(float v) {
  v = fmaxf(v, permxor16<1>(v));
  v = fmaxf(v, permxor16<2>(v));
  v = fmaxf(v, permxor16<4>(v));
  v = fmaxf(v, permxor16<8>(v));
  return v;
}
__device__ __forceinline__ float red_add32(float v) {
  v = red_add16(v);
  return v + permswap_half(v);
}

// ---------------------------------------------------------------------------
// Tensor Data Mover: DMA a 64x32 bf16 tile (row-major, leading dim `ld`) into
// LDS at byte offset lds_off, inserting 16B of padding after every 64B row so
// the LDS row stride is 40 bf16 (bank-spread, 16B-aligned). D# layout per
// CDNA5 ISA. Caller must guarantee full tiles.
// ---------------------------------------------------------------------------
#if ATHENA_TDM
__device__ __forceinline__ void tdm_load_tile_bf16(const bf16* gptr,
                                                   unsigned lds_off, int ld,
                                                   int rows_total) {
  unsigned long long ga = (unsigned long long)(size_t)gptr;
  u32x4 g0;
  i32x8 g1;
  i32x4 g2, g3;
  g0[0] = 1u;                                   // count=1, user descriptor
  g0[1] = lds_off;                              // lds_addr
  g0[2] = (unsigned)(ga & 0xFFFFFFFFu);         // global_addr[31:0]
  g0[3] = (unsigned)((ga >> 32) & 0x1FFFFFFu)   // global_addr[56:32]
        | (2u << 30);                           // type=2 ("image")
  g1[0] = (int)((1u << 16)                      // data_size = 2 bytes
              | (1u << 20)                      // pad_enable
              | (3u << 22)                      // pad_interval: 16 dwords
              | (3u << 25));                    // pad_amount: 4 dwords
  g1[1] = (int)(((unsigned)ld & 0xFFFFu) << 16);            // tensor_dim0 lo
  g1[2] = (int)((((unsigned)ld >> 16) & 0xFFFFu)            // tensor_dim0 hi
              | (((unsigned)rows_total & 0xFFFFu) << 16));  // tensor_dim1 lo
  g1[3] = (int)((((unsigned)rows_total >> 16) & 0xFFFFu)    // tensor_dim1 hi
              | (32u << 16));                               // tile_dim0 = 32
  g1[4] = 64;                                   // tile_dim1 = 64 rows
  g1[5] = ld;                                   // tensor_dim0_stride[31:0]
  g1[6] = 0;
  g1[7] = 0;
  g2[0] = g2[1] = g2[2] = g2[3] = 0;
  g3[0] = g3[1] = g3[2] = g3[3] = 0;
#if __clang_major__ >= 23
  i32x8 g4;
#pragma unroll
  for (int i = 0; i < 8; ++i) g4[i] = 0;
  __builtin_amdgcn_tensor_load_to_lds(g0, g1, g2, g3, g4, 0);
#else
  __builtin_amdgcn_tensor_load_to_lds(g0, g1, g2, g3, 0);
#endif
}
#endif

// ---------------------------------------------------------------------------
// Generic GEMM: C[M x Nout] = op(A[M x K] @ W[Nout x K]^T + bias)
// bf16 WMMA, f32 accumulate. 4 waves, 64x64 block tile, each wave 32x32.
// TDMA=true: A tiles are DMA'd by the Tensor Data Mover (requires bf16 A and
// M%64==0, K%32==0 so every tile is full).
// ---------------------------------------------------------------------------
template <typename AT, typename OT, bool RELU, bool TDMA = false>
__global__ void k_gemm(const AT* __restrict__ A, int lda,
                       const float* __restrict__ W, int ldw,
                       const float* __restrict__ bias,
                       OT* __restrict__ C, int ldc,
                       int M, int K, int Nout) {
  constexpr bool USE_TDM_A = TDMA && (ATHENA_TDM != 0);
  constexpr int AST = USE_TDM_A ? 40 : 72;   // LDS row stride (bf16 elements)
  __shared__ alignas(16) bf16 As[64][AST];
  __shared__ alignas(16) bf16 Bs[64][72];
  const int tid  = threadIdx.x;
  const int lane = tid & 31, wid = tid >> 5;
  const int wr = wid >> 1, wc = wid & 1;
  const int half = lane >> 4, l15 = lane & 15;
  const int m0 = blockIdx.x * 64, n0 = blockIdx.y * 64;

  v8f acc[2][2];
#pragma unroll
  for (int i = 0; i < 2; ++i)
#pragma unroll
    for (int j = 0; j < 2; ++j) acc[i][j] = v8f_zero();

  for (int k0 = 0; k0 < K; k0 += 32) {
    if constexpr (USE_TDM_A) {
#if ATHENA_TDM
      if (wid == 0)
        tdm_load_tile_bf16((const bf16*)A + (size_t)m0 * lda + k0,
                           (unsigned)(size_t)&As[0][0], lda, M);
#endif
    } else {
      // A staging: 8-element chunks, vectorized when fully in-bounds.
      for (int cid = tid; cid < 256; cid += 128) {
        int r = cid >> 2, c8 = (cid & 3) * 8;
        int gm = m0 + r, gk = k0 + c8;
        bf16* dst = &As[r][c8];
        if (gm < M && gk + 8 <= K) {
          if constexpr (__is_same(AT, bf16)) {
            *(uint4*)dst = *(const uint4*)((const bf16*)A + (size_t)gm * lda + gk);
          } else {
            const float* src = (const float*)A + (size_t)gm * lda + gk;
            float4 f0 = *(const float4*)src;
            float4 f1 = *(const float4*)(src + 4);
            union { bf16 h[8]; uint4 q; } t;
            t.h[0] = (bf16)f0.x; t.h[1] = (bf16)f0.y;
            t.h[2] = (bf16)f0.z; t.h[3] = (bf16)f0.w;
            t.h[4] = (bf16)f1.x; t.h[5] = (bf16)f1.y;
            t.h[6] = (bf16)f1.z; t.h[7] = (bf16)f1.w;
            *(uint4*)dst = t.q;
          }
        } else {
#pragma unroll
          for (int c = 0; c < 8; ++c) {
            int gkc = gk + c;
            float av = (gm < M && gkc < K) ? (float)A[(size_t)gm * lda + gkc] : 0.f;
            dst[c] = (bf16)av;
          }
        }
      }
    }
    // W staging (always f32 -> bf16).
    for (int cid = tid; cid < 256; cid += 128) {
      int r = cid >> 2, c8 = (cid & 3) * 8;
      int gn = n0 + r, gk = k0 + c8;
      bf16* dst = &Bs[r][c8];
      if (gn < Nout && gk + 8 <= K) {
        const float* src = W + (size_t)gn * ldw + gk;
        float4 f0 = *(const float4*)src;
        float4 f1 = *(const float4*)(src + 4);
        union { bf16 h[8]; uint4 q; } t;
        t.h[0] = (bf16)f0.x; t.h[1] = (bf16)f0.y;
        t.h[2] = (bf16)f0.z; t.h[3] = (bf16)f0.w;
        t.h[4] = (bf16)f1.x; t.h[5] = (bf16)f1.y;
        t.h[6] = (bf16)f1.z; t.h[7] = (bf16)f1.w;
        *(uint4*)dst = t.q;
      } else {
#pragma unroll
        for (int c = 0; c < 8; ++c) {
          int gkc = gk + c;
          float bv = (gn < Nout && gkc < K) ? W[(size_t)gn * ldw + gkc] : 0.f;
          dst[c] = (bf16)bv;
        }
      }
    }
    if constexpr (USE_TDM_A) {
#if ATHENA_TDM
      if (wid == 0) __builtin_amdgcn_s_wait_tensorcnt(0);
#endif
    }
    __syncthreads();
#pragma unroll
    for (int tr = 0; tr < 2; ++tr) {
      v16bf a = ld_frag(&As[32 * wr + 16 * tr + l15][0], half);
#pragma unroll
      for (int tc = 0; tc < 2; ++tc) {
        v16bf b = ld_frag(&Bs[32 * wc + 16 * tc + l15][0], half);
        acc[tr][tc] = wmma_bf16(a, b, acc[tr][tc]);
      }
    }
    __syncthreads();
  }

  // Epilogue: C layout -> lane = col (l15), element r = row r + 8*half.
#pragma unroll
  for (int tr = 0; tr < 2; ++tr)
#pragma unroll
    for (int tc = 0; tc < 2; ++tc)
#pragma unroll
      for (int r = 0; r < 8; ++r) {
        int grow = m0 + 32 * wr + 16 * tr + r + 8 * half;
        int gcol = n0 + 32 * wc + 16 * tc + l15;
        if (grow < M && gcol < Nout) {
          float v = acc[tr][tc][r] + bias[gcol];
          if (RELU) v = fmaxf(v, 0.f);
          C[(size_t)grow * ldc + gcol] = (OT)v;
        }
      }
}

// ---------------------------------------------------------------------------
// Flash attention: block = (qblock of 64 rows, head, n). 4 waves x 16 q-rows.
// Skv must be a multiple of 32 (512 self / 64 cross). DH=75 padded to 96.
// ---------------------------------------------------------------------------
__global__ void k_attn(const bf16* __restrict__ q, int q_rs,
                       const bf16* __restrict__ kv, int kv_rs,
                       int k_coloff, int v_coloff, int Skv,
                       bf16* __restrict__ out, int out_rs) {
  __shared__ alignas(16) bf16 Qs[64][104];   // 64 q-rows x 96 dh (padded)
  __shared__ alignas(16) bf16 Ks[32][104];   // 32 keys  x 96 dh
  __shared__ alignas(16) bf16 Vt[96][40];    // dh-major V (keys in row)
  __shared__ alignas(16) bf16 Ps[4][16][40]; // per-wave P tile 16x32
  const int tid = threadIdx.x, lane = tid & 31, wid = tid >> 5;
  const int half = lane >> 4, l15 = lane & 15;
  const int n = blockIdx.z, h = blockIdx.y, qb = blockIdx.x;
  const float scale = 0.11547005383792516f;  // 1/sqrt(75)

  for (int idx = tid; idx < 64 * 96; idx += 128) {
    int r = idx / 96, c = idx - 96 * r;
    float v = 0.f;
    if (c < 75) v = (float)q[((size_t)n * 512 + qb * 64 + r) * q_rs + h * 75 + c];
    Qs[r][c] = (bf16)v;
  }

  v8f O[5];
#pragma unroll
  for (int j = 0; j < 5; ++j) O[j] = v8f_zero();
  float mr[8], lsum[8];
#pragma unroll
  for (int r = 0; r < 8; ++r) { mr[r] = -__builtin_inff(); lsum[r] = 0.f; }
  __syncthreads();

  for (int key0 = 0; key0 < Skv; key0 += 32) {
    for (int idx = tid; idx < 32 * 96; idx += 128) {
      int r = idx / 96, c = idx - 96 * r;
      float kvv = 0.f, vvv = 0.f;
      if (c < 75) {
        size_t rowb = ((size_t)n * Skv + key0 + r) * kv_rs;
        kvv = (float)kv[rowb + k_coloff + h * 75 + c];
        vvv = (float)kv[rowb + v_coloff + h * 75 + c];
      }
      Ks[r][c] = (bf16)kvv;
      Vt[c][r] = (bf16)vvv;
    }
    __syncthreads();

    // S tiles: 16 q-rows x 32 keys (two 16-wide C tiles)
    v8f s0 = v8f_zero(), s1 = v8f_zero();
#pragma unroll
    for (int kk = 0; kk < 3; ++kk) {
      v16bf a  = ld_frag(&Qs[16 * wid + l15][32 * kk], half);
      v16bf b0 = ld_frag(&Ks[l15][32 * kk], half);
      v16bf b1 = ld_frag(&Ks[16 + l15][32 * kk], half);
      s0 = wmma_bf16(a, b0, s0);
      s1 = wmma_bf16(a, b1, s1);
    }

    // Online softmax: row of element r is (r + 8*half); row reduction is a
    // 16-lane butterfly on the VALU (permlane16), aligned with the C layout.
#pragma unroll
    for (int r = 0; r < 8; ++r) {
      float a0 = s0[r] * scale, a1 = s1[r] * scale;
      float t = red_max16(fmaxf(a0, a1));
      float nm = fmaxf(mr[r], t);
      float alpha = __expf(mr[r] - nm);
      float p0 = __expf(a0 - nm), p1 = __expf(a1 - nm);
      float rs = red_add16(p0 + p1);
      lsum[r] = lsum[r] * alpha + rs;
      mr[r] = nm;
#pragma unroll
      for (int j = 0; j < 5; ++j) O[j][r] *= alpha;
      Ps[wid][r + 8 * half][l15]      = (bf16)p0;
      Ps[wid][r + 8 * half][16 + l15] = (bf16)p1;
    }
    asm volatile("s_wait_dscnt 0" ::: "memory");  // per-wave LDS RAW

    // O += P(16x32) @ V(32x80): A from Ps, B from dh-major Vt.
    v16bf pa = ld_frag(&Ps[wid][l15][0], half);
#pragma unroll
    for (int j = 0; j < 5; ++j) {
      v16bf vb = ld_frag(&Vt[16 * j + l15][0], half);
      O[j] = wmma_bf16(pa, vb, O[j]);
    }
    __syncthreads();
  }

#pragma unroll
  for (int r = 0; r < 8; ++r) {
    float inv = 1.f / lsum[r];
    int grow = qb * 64 + wid * 16 + r + 8 * half;
    size_t rowb = ((size_t)n * 512 + grow) * out_rs + h * 75;
#pragma unroll
    for (int j = 0; j < 5; ++j) {
      int c = 16 * j + l15;
      if (c < 75) out[rowb + c] = (bf16)(O[j][r] * inv);
    }
  }
}

// ---------------------------------------------------------------------------
// LayerNorm with residual: out_row = LN(x_row + y_row) * g + b.  One wave/row.
// ---------------------------------------------------------------------------
__global__ void k_ln(const float* __restrict__ x, const float* __restrict__ y,
                     float* __restrict__ out, const float* __restrict__ g,
                     const float* __restrict__ b, int rows) {
  int wid = threadIdx.x >> 5, lane = threadIdx.x & 31;
  int row = blockIdx.x * 8 + wid;
  if (row >= rows) return;
  const float* xr = x + (size_t)row * 300;
  const float* yr = y + (size_t)row * 300;
  float vv[10];
  float s = 0.f;
  int i = 0;
  for (int d = lane; d < 300; d += 32, ++i) {
    float v = xr[d] + yr[d];
    vv[i] = v;
    s += v;
  }
  s = red_add32(s);
  float mean = s / 300.f;
  float vs = 0.f;
  i = 0;
  for (int d = lane; d < 300; d += 32, ++i) {
    float c = vv[i] - mean;
    vs += c * c;
  }
  vs = red_add32(vs);
  float inv = rsqrtf(vs / 300.f + 1e-5f);
  float* orow = out + (size_t)row * 300;
  i = 0;
  for (int d = lane; d < 300; d += 32, ++i)
    orow[d] = (vv[i] - mean) * inv * g[d] + b[d];
}

// ---------------------------------------------------------------------------
// Candidate scores, targets, validity. One block per n.
// ---------------------------------------------------------------------------
__global__ void k_scores(const float* __restrict__ cde,   // (64,30,300)
                         const float* __restrict__ m,     // (64,300)
                         const float* __restrict__ cet,   // (64,30)
                         const int* __restrict__ span_st,
                         const int* __restrict__ span_ln,
                         float* __restrict__ scores,      // (64,31)
                         int* __restrict__ targets, int* __restrict__ validf,
                         int* __restrict__ cntarr, int* __restrict__ tgtidx) {
  __shared__ float sc[31];
  int n = blockIdx.x, tid = threadIdx.x, lane = tid & 31, wid = tid >> 5;
  for (int c = wid; c < 30; c += 4) {
    const float* row = cde + ((size_t)n * 30 + c) * 300;
    float p = 0.f;
    for (int d = lane; d < 300; d += 32) p += row[d] * m[(size_t)n * 300 + d];
    p = red_add32(p);
    if (lane == 0) {
      bool mult = row[0] != 0.f;
      sc[c] = mult ? p : -100.0f;
    }
  }
  if (tid == 0) sc[30] = 0.f;
  __syncthreads();
  if (tid == 0) {
    int tg = 0;
    float bv = cet[(size_t)n * 30];
    for (int c = 1; c < 30; ++c) {
      float v = cet[(size_t)n * 30 + c];
      if (v > bv) { bv = v; tg = c; }
    }
    if (sc[tg] == -100.0f) tg = 30;
    int start = span_st[n], len = span_ln[n];
    int lo = start < 0 ? 0 : start;
    int hi = start + len; if (hi > 512) hi = 512;
    int cnt = hi - lo; if (cnt < 0) cnt = 0;
    bool valid = (len > 0) && (tg < 30) && (cnt > 0);
    targets[n] = tg;
    validf[n] = valid ? 1 : 0;
    cntarr[n] = cnt;
    int ti = tg; if (ti > 29) ti = 29; if (ti < 0) ti = 0;
    tgtidx[n] = valid ? ti : 0;
  }
  __syncthreads();
  if (tid < 31) scores[(size_t)n * 31 + tid] = sc[tid];
}

// ---------------------------------------------------------------------------
// Gathers: x[n] = seq[batch_ids[n]], mem[n] = cse[n, tgt_idx[n]].
// grid = (5, 64): chunk 0 -> mem, chunks 1..4 -> 128 seq rows each.
// ---------------------------------------------------------------------------
__global__ void k_gather(const float* __restrict__ seq,
                         const float* __restrict__ cse,
                         const int* __restrict__ batch_ids,
                         const int* __restrict__ tgtidx,
                         float* __restrict__ x, float* __restrict__ mem) {
  int n = blockIdx.y, chunk = blockIdx.x, tid = threadIdx.x;
  if (chunk == 0) {
    int ti = tgtidx[n];
    const float* src = cse + ((size_t)n * 30 + ti) * 64 * 300;
    float* dst = mem + (size_t)n * 64 * 300;
    for (int idx = tid; idx < 64 * 300; idx += 256) dst[idx] = src[idx];
  } else {
    int b = batch_ids[n];
    int s0 = (chunk - 1) * 128;
    const float* src = seq + ((size_t)b * 512 + s0) * 300;
    float* dst = x + ((size_t)n * 512 + s0) * 300;
    for (int idx = tid; idx < 128 * 300; idx += 256) dst[idx] = src[idx];
  }
}

// ---------------------------------------------------------------------------
// Span average, rescore, per-row softmax + per-row loss. One block per n.
// ---------------------------------------------------------------------------
__global__ void k_final(const float* __restrict__ x,    // (64,512,300)
                        const float* __restrict__ cse,  // (64,30,64,300)
                        const float* __restrict__ scores,
                        const int* __restrict__ targets,
                        const int* __restrict__ validf,
                        const int* __restrict__ cntarr,
                        const int* __restrict__ span_st,
                        const int* __restrict__ span_ln,
                        float* __restrict__ out,        // [1 + 64*31]
                        float* __restrict__ lossp) {
  __shared__ float avgs[300];
  __shared__ float news[30];
  __shared__ float fins[31];
  __shared__ float sh_logZ;
  int n = blockIdx.x, tid = threadIdx.x, lane = tid & 31, wid = tid >> 5;
  int cnt = cntarr[n], start = span_st[n], len = span_ln[n];
  int lo = start < 0 ? 0 : start;
  int hi = start + len; if (hi > 512) hi = 512;
  float inv = 1.f / (float)(cnt > 0 ? cnt : 1);
  for (int d = tid; d < 300; d += 128) {
    float s = 0.f;
    for (int r = lo; r < hi; ++r) s += x[((size_t)n * 512 + r) * 300 + d];
    avgs[d] = s * inv;
  }
  __syncthreads();
  for (int c = wid; c < 30; c += 4) {
    const float* row = cse + ((size_t)n * 30 + c) * (size_t)64 * 300;  // l = 0
    float p = 0.f;
    for (int d = lane; d < 300; d += 32) p += row[d] * avgs[d];
    p = red_add32(p);
    if (lane == 0) news[c] = p;
  }
  __syncthreads();
  int valid = validf[n];
  if (tid < 31)
    fins[tid] = (tid < 30 && valid) ? news[tid] : scores[(size_t)n * 31 + tid];
  __syncthreads();
  if (tid == 0) {
    float mx = fins[0];
    for (int c = 1; c < 31; ++c) mx = fmaxf(mx, fins[c]);
    float Z = 0.f;
    for (int c = 0; c < 31; ++c) Z += __expf(fins[c] - mx);
    float logZ = mx + __logf(Z);
    sh_logZ = logZ;
    lossp[n] = -(fins[targets[n]] - logZ);
  }
  __syncthreads();
  if (tid < 31) out[1 + n * 31 + tid] = __expf(fins[tid] - sh_logZ);
}

__global__ void k_loss(const float* __restrict__ lossp, float* __restrict__ out) {
  if (threadIdx.x == 0 && blockIdx.x == 0) {
    float s = 0.f;
    for (int i = 0; i < 64; ++i) s += lossp[i];  // fixed order -> deterministic
    out[0] = s / 64.f;
  }
}

// ---------------------------------------------------------------------------
extern "C" void kernel_launch(void* const* d_in, const int* in_sizes, int n_in,
                              void* d_out, int out_size, void* d_ws, size_t ws_size,
                              hipStream_t stream) {
  (void)in_sizes; (void)n_in; (void)out_size; (void)ws_size;
  const float* mention   = (const float*)d_in[0];
  const float* seq_emb   = (const float*)d_in[1];
  const float* cde       = (const float*)d_in[2];
  const float* cse       = (const float*)d_in[3];
  const float* cet       = (const float*)d_in[4];
  const int*   span_st   = (const int*)d_in[5];
  const int*   span_ln   = (const int*)d_in[6];
  const int*   batch_ids = (const int*)d_in[7];
  const float* Wm        = (const float*)d_in[8];
  const float* bm        = (const float*)d_in[9];
  const float* sa_in_w   = (const float*)d_in[10];
  const float* sa_in_b   = (const float*)d_in[11];
  const float* sa_out_w  = (const float*)d_in[12];
  const float* sa_out_b  = (const float*)d_in[13];
  const float* ca_in_w   = (const float*)d_in[14];
  const float* ca_in_b   = (const float*)d_in[15];
  const float* ca_out_w  = (const float*)d_in[16];
  const float* ca_out_b  = (const float*)d_in[17];
  const float* ff1_w     = (const float*)d_in[18];
  const float* ff1_b     = (const float*)d_in[19];
  const float* ff2_w     = (const float*)d_in[20];
  const float* ff2_b     = (const float*)d_in[21];
  const float* ln1_g     = (const float*)d_in[22];
  const float* ln1_b     = (const float*)d_in[23];
  const float* ln2_g     = (const float*)d_in[24];
  const float* ln2_b     = (const float*)d_in[25];
  const float* ln3_g     = (const float*)d_in[26];
  const float* ln3_b     = (const float*)d_in[27];

  char* ws = (char*)d_ws;
  size_t off = 0;
  auto alloc = [&](size_t bytes) -> char* {
    char* p = ws + off;
    off = (off + bytes + 255) & ~(size_t)255;
    return p;
  };
  float* seq     = (float*)alloc(4096ull * 300 * 4);
  float* x       = (float*)alloc(32768ull * 300 * 4);
  float* mbuf    = (float*)alloc(64ull * 300 * 4);
  float* mem     = (float*)alloc(64ull * 64 * 300 * 4);
  float* scores  = (float*)alloc(64 * 31 * 4);
  int*   targets = (int*)alloc(64 * 4);
  int*   validf  = (int*)alloc(64 * 4);
  int*   cntarr  = (int*)alloc(64 * 4);
  int*   tgtidx  = (int*)alloc(64 * 4);
  float* lossp   = (float*)alloc(64 * 4);
  bf16*  qkv     = (bf16*)alloc(32768ull * 900 * 2);
  bf16*  attnout = (bf16*)alloc(32768ull * 300 * 2);
  float* proj    = (float*)alloc(32768ull * 300 * 4);
  bf16*  hbuf    = (bf16*)alloc(8192ull * 2048 * 2);
  bf16*  caq     = qkv;                   // self-attn qkv dead by then: reuse
  bf16*  cakv    = qkv + 32768ull * 300;

  dim3 blk(128);
  // m, seq projections (K=768)
  k_gemm<float, float, false><<<dim3(1, 5), blk, 0, stream>>>(
      mention, 768, Wm, 768, bm, mbuf, 300, 64, 768, 300);
  k_gemm<float, float, false><<<dim3(64, 5), blk, 0, stream>>>(
      seq_emb, 768, Wm, 768, bm, seq, 300, 4096, 768, 300);
  // candidate scores + targets + validity
  k_scores<<<dim3(64), blk, 0, stream>>>(cde, mbuf, cet, span_st, span_ln,
                                         scores, targets, validf, cntarr, tgtidx);
  // gathers: x = seq[batch_ids], mem = cse[n, tgt_idx]
  k_gather<<<dim3(5, 64), dim3(256), 0, stream>>>(seq, cse, batch_ids, tgtidx, x, mem);
  // self-attn: fused QKV, flash attention, out-proj, LN1
  k_gemm<float, bf16, false><<<dim3(512, 15), blk, 0, stream>>>(
      x, 300, sa_in_w, 300, sa_in_b, qkv, 900, 32768, 300, 900);
  k_attn<<<dim3(8, 4, 64), blk, 0, stream>>>(qkv, 900, qkv, 900, 300, 600, 512,
                                             attnout, 300);
  k_gemm<bf16, float, false><<<dim3(512, 5), blk, 0, stream>>>(
      attnout, 300, sa_out_w, 300, sa_out_b, proj, 300, 32768, 300, 300);
  k_ln<<<dim3(4096), dim3(256), 0, stream>>>(x, proj, x, ln1_g, ln1_b, 32768);
  // cross-attn: q from x (Wq rows), k/v from mem (Wk|Wv rows), LN2
  k_gemm<float, bf16, false><<<dim3(512, 5), blk, 0, stream>>>(
      x, 300, ca_in_w, 300, ca_in_b, caq, 300, 32768, 300, 300);
  k_gemm<float, bf16, false><<<dim3(64, 10), blk, 0, stream>>>(
      mem, 300, ca_in_w + 300 * 300, 300, ca_in_b + 300, cakv, 600, 4096, 300, 600);
  k_attn<<<dim3(8, 4, 64), blk, 0, stream>>>(caq, 300, cakv, 600, 0, 300, 64,
                                             attnout, 300);
  k_gemm<bf16, float, false><<<dim3(512, 5), blk, 0, stream>>>(
      attnout, 300, ca_out_w, 300, ca_out_b, proj, 300, 32768, 300, 300);
  k_ln<<<dim3(4096), dim3(256), 0, stream>>>(x, proj, x, ln2_g, ln2_b, 32768);
  // FFN in 4 row-chunks (bf16 intermediate reused), LN3.
  // FFN-2 is fully tiled (M=8192, K=2048) -> A tiles via Tensor Data Mover.
  for (int c = 0; c < 4; ++c) {
    const float* Ax = x + (size_t)c * 8192 * 300;
    float* Px = proj + (size_t)c * 8192 * 300;
    k_gemm<float, bf16, true><<<dim3(128, 32), blk, 0, stream>>>(
        Ax, 300, ff1_w, 300, ff1_b, hbuf, 2048, 8192, 300, 2048);
    k_gemm<bf16, float, false, true><<<dim3(128, 5), blk, 0, stream>>>(
        hbuf, 2048, ff2_w, 2048, ff2_b, Px, 300, 8192, 2048, 300);
  }
  k_ln<<<dim3(4096), dim3(256), 0, stream>>>(x, proj, x, ln3_g, ln3_b, 32768);
  // span average, rescoring, softmax, loss
  k_final<<<dim3(64), blk, 0, stream>>>(x, cse, scores, targets, validf, cntarr,
                                        span_st, span_ln, (float*)d_out, lossp);
  k_loss<<<dim3(1), dim3(32), 0, stream>>>(lossp, (float*)d_out);
}
